// MultiHeadSelfAttention_74208444940465
// MI455X (gfx1250) — matrix operand order repaired and checked by reading, not compiled, not run on hardware
//
#include <hip/hip_runtime.h>
#include <hip/hip_bf16.h>

// ---------------------------------------------------------------------------
// MI455X (gfx1250) multi-head self-attention forward.
// B=4, S=2048, D=1024, H=16, DK=64, causal, RoPE theta=10000.
// Matrix math via v_wmma_f32_16x16x32_bf16 (wave32); GEMM weight tiles staged
// into LDS by the Tensor Data Mover (tensor_load_to_lds + s_wait_tensorcnt),
// double-buffered so the DMA overlaps the WMMA stream.
// ---------------------------------------------------------------------------

typedef __bf16 bf16_t;
typedef __attribute__((ext_vector_type(16))) __bf16 v16bf;
typedef __attribute__((ext_vector_type(8)))  float  v8f;
typedef unsigned int v4u __attribute__((ext_vector_type(4)));
typedef unsigned int v8u __attribute__((ext_vector_type(8)));

#define NB   4
#define NS   2048
#define ND   1024
#define NH   16
#define NDK  64
#define N3D  3072
#define LOG_THETA 9.210340371976184f   // ln(10000)

union Frag {
    v16bf v;
    struct { uint4 lo; uint4 hi; } q;
};

__device__ __forceinline__ uint4 ldg16(const bf16_t* p) {
    return *reinterpret_cast<const uint4*>(p);
}

// Plain 16-byte LDS read -> ds_load_b128 after addrspace inference. Kept live
// across iterations because the tile buffer pointer escapes into the TDM asm
// (whose "memory" clobber means "DMA may write this").
__device__ __forceinline__ uint4 ldlds16(const bf16_t* p) {
    return *reinterpret_cast<const uint4*>(p);
}

__device__ __forceinline__ v8f wmma_bf16(v16bf a, v16bf b, v8f c) {
    // D = A(16x32 bf16) * B(32x16 bf16) + C(16x16 f32)
    return __builtin_amdgcn_wmma_f32_16x16x32_bf16(
        /*neg_a=*/false, a, /*neg_b=*/false, b,
        /*c_mod=*/(short)0, c, /*reuse_a=*/false, /*reuse_b=*/false);
}

// ---------------------------------------------------------------------------
// Tensor Data Mover: 2D tile load Global -> LDS (D# groups 0/1, 2-group form).
// data_size = 2 bytes (bf16). Issued once per wave (EXEC ignored by TDM).
//   group0: [0]=count=1  [1]=lds_addr  [2]=gaddr_lo  [3]=gaddr_hi | type=2<<30
//   group1: [0]=data_size<<16  [1..3]=tensor_dim0/1 + tile_dim0
//           [4]=tile_dim1      [5]=tensor_dim0_stride (elements)
// The LDS destination pointer is ALSO passed as an asm operand so it escapes:
// the "memory" clobber then forces the compiler to treat the buffer as
// written by the DMA (otherwise LDS reads of it get folded away).
// Generic pointers to LDS = {shared aperture, lds byte offset} (ISA 10.2),
// so the low 32 bits are exactly the descriptor's lds_addr.
// ---------------------------------------------------------------------------
__device__ __forceinline__ void tdm_load_tile_2d(const void* ldsp,
                                                 const bf16_t* gptr,
                                                 unsigned tensor_d0,
                                                 unsigned tensor_d1,
                                                 unsigned stride0_elems,
                                                 unsigned tile_d0,
                                                 unsigned tile_d1) {
    unsigned long long ga = (unsigned long long)(uintptr_t)gptr;
    v4u g0;
    g0[0] = 1u;                                   // count=1, user descriptor
    g0[1] = (unsigned)(uintptr_t)ldsp;            // LDS byte offset
    g0[2] = (unsigned)ga;                         // global_addr[31:0]
    g0[3] = (unsigned)(ga >> 32) | 0x80000000u;   // global_addr[56:32] | type=2
    v8u g1;
    g1[0] = 1u << 16;                             // data_size=1 -> 2 bytes
    g1[1] = (tensor_d0 & 0xFFFFu) << 16;          // tensor_dim0[15:0]
    g1[2] = (tensor_d0 >> 16) | ((tensor_d1 & 0xFFFFu) << 16);
    g1[3] = (tensor_d1 >> 16) | (tile_d0 << 16);  // tile_dim0
    g1[4] = tile_d1 & 0xFFFFu;                    // tile_dim1 (tile_dim2=0)
    g1[5] = stride0_elems;                        // tensor_dim0_stride low32
    g1[6] = 0u;
    g1[7] = 0u;
    asm volatile("tensor_load_to_lds %0, %1"
                 :: "s"(g0), "s"(g1), "s"(ldsp)   // %2 unreferenced: escapes LDS
                 : "memory");
}

// ---------------------------------------------------------------------------
// Stage 0: f32 -> bf16 conversion (halves GEMM read bandwidth).
// ---------------------------------------------------------------------------
__global__ void cvt_f32_bf16(const float* __restrict__ in,
                             bf16_t* __restrict__ out, int n) {
    int i = blockIdx.x * blockDim.x + threadIdx.x;
    int stride = gridDim.x * blockDim.x;
    for (; i < n; i += stride) out[i] = (bf16_t)in[i];
}

// ---------------------------------------------------------------------------
// Stage 1: QKV GEMM + bias + RoPE + scatter into attention-friendly layouts.
//   qkv[m][e] = sum_k xb[m][k] * wb[e][k] + bias[e]
//   Q,K -> [B*H][S][DK] ; V -> [B*H][DK][S] (transposed)
// Block = 128 threads (4 waves); block tile 64(M) x 32(N); weight tile
// (32 N-rows x 32 K) TDM-staged into double-buffered LDS, shared by all waves.
// ---------------------------------------------------------------------------
__global__ void qkv_rope_kernel(const bf16_t* __restrict__ xb,   // [B*S][D]
                                const bf16_t* __restrict__ wb,   // [3D][D]
                                const float*  __restrict__ bias, // [3D]
                                const int*    __restrict__ tpos, // [S]
                                bf16_t* __restrict__ Qb,
                                bf16_t* __restrict__ Kb,
                                bf16_t* __restrict__ Vt) {
    __shared__ __align__(16) bf16_t tileB[2][32][32];

    const int lane  = threadIdx.x & 31;
    const int wave  = threadIdx.x >> 5;
    const int nbase = blockIdx.x * 32;
    const int mbase = blockIdx.y * 64 + wave * 16;

    const int mrow  = lane & 15;   // A: row per lane / B: col per lane
    const int khalf = lane >> 4;
    const int akoff = khalf * 16;  // A-frag: 16 contiguous K per lane
    const int bkoff = khalf * 16;  // B-frag: 16 contiguous K per lane

    const bf16_t* arow = xb + (size_t)(mbase + mrow) * ND;
    const bf16_t* wrow = wb + (size_t)nbase * ND;   // tile origin row

    v8f acc0 = {0,0,0,0,0,0,0,0};
    v8f acc1 = {0,0,0,0,0,0,0,0};

    const int iters = ND / 32;
    if (wave == 0)
        tdm_load_tile_2d(&tileB[0][0][0], wrow, ND, 32, ND, 32, 32);

    for (int it = 0; it < iters; ++it) {
        const int k = it * 32;
        if (wave == 0) {
            if (it + 1 < iters) {
                tdm_load_tile_2d(&tileB[(it + 1) & 1][0][0], wrow + (k + 32),
                                 ND, 32, ND, 32, 32);
                __builtin_amdgcn_s_wait_tensorcnt(1);   // tile `it` landed
            } else {
                __builtin_amdgcn_s_wait_tensorcnt(0);
            }
        }
        __syncthreads();

        const int buf = it & 1;
        Frag a, b0, b1;
        a.q.lo  = ldg16(arow + k + akoff);
        a.q.hi  = ldg16(arow + k + akoff + 8);
        b0.q.lo = ldlds16(&tileB[buf][mrow][bkoff]);
        b0.q.hi = ldlds16(&tileB[buf][mrow][bkoff + 8]);
        b1.q.lo = ldlds16(&tileB[buf][16 + mrow][bkoff]);
        b1.q.hi = ldlds16(&tileB[buf][16 + mrow][bkoff + 8]);
        acc0 = wmma_bf16(a.v, b0.v, acc0);
        acc1 = wmma_bf16(a.v, b1.v, acc1);

        __syncthreads();
    }

    // Epilogue: C frag: VGPR r -> row (mbase + r + 8*khalf), col = lane&15.
    #pragma unroll
    for (int cf = 0; cf < 2; ++cf) {
        v8f acc = cf ? acc1 : acc0;
        const int e    = nbase + cf * 16 + (lane & 15);
        const int sec  = e >> 10;       // 0=Q 1=K 2=V  (uniform across wave)
        const int col  = e & 1023;
        const int head = col >> 6;
        const int dk   = col & 63;
        const float bv = bias[e];
        #pragma unroll
        for (int r = 0; r < 8; ++r) {
            const int row  = mbase + r + 8 * khalf;
            const int s    = row & (NS - 1);
            const int bidx = row >> 11;            // row / S
            const size_t bh = (size_t)(bidx * NH + head);
            float val = acc[r] + bv;
            if (sec == 2) {
                Vt[(bh * NDK + dk) * NS + s] = (bf16_t)val;
            } else {
                const int j = dk >> 1;
                const float inv_freq = __expf(-((float)(2 * j) / (float)NDK) * LOG_THETA);
                const float ang = (float)tpos[s] * inv_freq;
                float sn, cs;
                __sincosf(ang, &sn, &cs);
                const float other = __shfl_xor(val, 1, 32);   // RoPE pair lane
                const float res = (dk & 1) ? (other * sn + val * cs)
                                           : (val * cs - other * sn);
                bf16_t* dst = (sec == 0) ? Qb : Kb;
                dst[(bh * NS + s) * NDK + dk] = (bf16_t)res;
            }
        }
    }
}

// ---------------------------------------------------------------------------
// Stage 2: causal flash attention. Block = 128 (4 waves), wave = 16 q rows.
// Per 32-key tile: 4 WMMA for S = Q·K^T, online softmax, 4 WMMA for P·V.
// P (C-layout) -> A-layout transposed through per-wave LDS.
// ---------------------------------------------------------------------------
__global__ void flash_attn_kernel(const bf16_t* __restrict__ Qb,
                                  const bf16_t* __restrict__ Kb,
                                  const bf16_t* __restrict__ Vt,
                                  bf16_t* __restrict__ AO) {   // [B*S][D]
    __shared__ __align__(16) bf16_t plds[4][16][32];

    const int lane = threadIdx.x & 31;
    const int wave = threadIdx.x >> 5;
    const int bh   = blockIdx.y;                 // 0..B*H-1
    const int qb   = blockIdx.x * 64 + wave * 16;

    const int mrow  = lane & 15;
    const int khalf = lane >> 4;

    // Q: two A fragments covering dk 0-31 and 32-63.
    const bf16_t* Qrow = Qb + ((size_t)bh * NS + qb + mrow) * NDK;
    Frag qf0, qf1;
    qf0.q.lo = ldg16(Qrow + khalf * 8);
    qf0.q.hi = ldg16(Qrow + khalf * 8 + 16);
    qf1.q.lo = ldg16(Qrow + 32 + khalf * 8);
    qf1.q.hi = ldg16(Qrow + 32 + khalf * 8 + 16);

    v8f o0 = {0,0,0,0,0,0,0,0}, o1 = {0,0,0,0,0,0,0,0};
    v8f o2 = {0,0,0,0,0,0,0,0}, o3 = {0,0,0,0,0,0,0,0};
    float mi[8], li[8];
    #pragma unroll
    for (int r = 0; r < 8; ++r) { mi[r] = -1e30f; li[r] = 0.0f; }

    const float scale = 0.125f;                  // 1/sqrt(DK)
    const int ktend = qb + 16;                   // causal bound

    for (int kt = 0; kt < ktend; kt += 32) {
        if (kt + 32 < ktend)                     // warm next K tile
            __builtin_prefetch(Kb + ((size_t)bh * NS + kt + 32 + mrow) * NDK, 0, 0);

        // ---- S = Q * K^T  (two 16x16 f32 frags, keys kt..kt+31)
        v8f s0 = {0,0,0,0,0,0,0,0}, s1 = {0,0,0,0,0,0,0,0};
        #pragma unroll
        for (int f = 0; f < 2; ++f) {
            const bf16_t* Krow =
                Kb + ((size_t)bh * NS + kt + 16 * f + mrow) * NDK + 8 * khalf;
            Frag k0, k1;                          // B frags: dk chunks 0-31 / 32-63
            k0.q.lo = ldg16(Krow);
            k0.q.hi = ldg16(Krow + 16);
            k1.q.lo = ldg16(Krow + 32);
            k1.q.hi = ldg16(Krow + 48);
            if (f == 0) { s0 = wmma_bf16(qf0.v, k0.v, s0); s0 = wmma_bf16(qf1.v, k1.v, s0); }
            else        { s1 = wmma_bf16(qf0.v, k0.v, s1); s1 = wmma_bf16(qf1.v, k1.v, s1); }
        }

        // ---- causal mask + scale + row max
        float rm[8];
        #pragma unroll
        for (int r = 0; r < 8; ++r) {
            const int q    = qb + r + 8 * khalf;
            const int key0 = kt + (lane & 15);
            float v0 = s0[r] * scale; if (key0 > q)      v0 = -1e9f;
            float v1 = s1[r] * scale; if (key0 + 16 > q) v1 = -1e9f;
            s0[r] = v0; s1[r] = v1;
            rm[r] = fmaxf(v0, v1);
        }
        #pragma unroll
        for (int off = 1; off < 16; off <<= 1) {
            #pragma unroll
            for (int r = 0; r < 8; ++r)
                rm[r] = fmaxf(rm[r], __shfl_xor(rm[r], off, 32));
        }

        // ---- online softmax update
        float alpha[8], rs[8];
        #pragma unroll
        for (int r = 0; r < 8; ++r) {
            const float nm = fmaxf(mi[r], rm[r]);
            alpha[r] = __expf(mi[r] - nm);
            mi[r] = nm;
            s0[r] = __expf(s0[r] - nm);
            s1[r] = __expf(s1[r] - nm);
            rs[r] = s0[r] + s1[r];
        }
        #pragma unroll
        for (int off = 1; off < 16; off <<= 1) {
            #pragma unroll
            for (int r = 0; r < 8; ++r)
                rs[r] += __shfl_xor(rs[r], off, 32);
        }
        #pragma unroll
        for (int r = 0; r < 8; ++r) {
            li[r] = li[r] * alpha[r] + rs[r];
            o0[r] *= alpha[r]; o1[r] *= alpha[r];
            o2[r] *= alpha[r]; o3[r] *= alpha[r];
        }

        // ---- P: C-layout -> A-layout via per-wave LDS transpose
        #pragma unroll
        for (int r = 0; r < 8; ++r) {
            plds[wave][r + 8 * khalf][lane & 15]        = (bf16_t)s0[r];
            plds[wave][r + 8 * khalf][16 + (lane & 15)] = (bf16_t)s1[r];
        }
        asm volatile("s_wait_dscnt 0" ::: "memory");
        Frag pf;
        pf.q.lo = *(const uint4*)&plds[wave][mrow][khalf * 8];
        pf.q.hi = *(const uint4*)&plds[wave][mrow][khalf * 8 + 16];

        // ---- O += P * V   (V stored transposed: contiguous in k)
        #pragma unroll
        for (int j = 0; j < 4; ++j) {
            const bf16_t* Vrow =
                Vt + ((size_t)bh * NDK + j * 16 + mrow) * NS + kt + 8 * khalf;
            Frag vf;
            vf.q.lo = ldg16(Vrow);
            vf.q.hi = ldg16(Vrow + 16);
            v8f& oj = (j == 0) ? o0 : (j == 1) ? o1 : (j == 2) ? o2 : o3;
            oj = wmma_bf16(pf.v, vf.v, oj);
        }
    }

    // ---- normalize and store into [B][S][D] (col = h*64 + dk)
    const int b = bh >> 4, h = bh & 15;
    #pragma unroll
    for (int r = 0; r < 8; ++r) {
        const int q = qb + r + 8 * khalf;
        const float inv = 1.0f / li[r];
        const size_t base = ((size_t)(b * NS + q)) * ND + h * NDK + (lane & 15);
        AO[base]      = (bf16_t)(o0[r] * inv);
        AO[base + 16] = (bf16_t)(o1[r] * inv);
        AO[base + 32] = (bf16_t)(o2[r] * inv);
        AO[base + 48] = (bf16_t)(o3[r] * inv);
    }
}

// ---------------------------------------------------------------------------
// Stage 3: output projection: out[m][n] = sum_k AO[m][k]*Wo[n][k] + bias[n]
// Same TDM-staged weight-tile scheme as stage 1.
// ---------------------------------------------------------------------------
__global__ void oproj_kernel(const bf16_t* __restrict__ AO,   // [B*S][D]
                             const bf16_t* __restrict__ wob,  // [D][D]
                             const float*  __restrict__ bias, // [D]
                             float* __restrict__ out) {       // [B*S][D]
    __shared__ __align__(16) bf16_t tileB[2][32][32];

    const int lane  = threadIdx.x & 31;
    const int wave  = threadIdx.x >> 5;
    const int nbase = blockIdx.x * 32;
    const int mbase = blockIdx.y * 64 + wave * 16;

    const int mrow  = lane & 15;
    const int khalf = lane >> 4;
    const int akoff = khalf * 16;
    const int bkoff = khalf * 16;

    const bf16_t* arow = AO  + (size_t)(mbase + mrow) * ND;
    const bf16_t* wrow = wob + (size_t)nbase * ND;

    v8f acc0 = {0,0,0,0,0,0,0,0};
    v8f acc1 = {0,0,0,0,0,0,0,0};

    const int iters = ND / 32;
    if (wave == 0)
        tdm_load_tile_2d(&tileB[0][0][0], wrow, ND, 32, ND, 32, 32);

    for (int it = 0; it < iters; ++it) {
        const int k = it * 32;
        if (wave == 0) {
            if (it + 1 < iters) {
                tdm_load_tile_2d(&tileB[(it + 1) & 1][0][0], wrow + (k + 32),
                                 ND, 32, ND, 32, 32);
                __builtin_amdgcn_s_wait_tensorcnt(1);
            } else {
                __builtin_amdgcn_s_wait_tensorcnt(0);
            }
        }
        __syncthreads();

        const int buf = it & 1;
        Frag a, b0, b1;
        a.q.lo  = ldg16(arow + k + akoff);
        a.q.hi  = ldg16(arow + k + akoff + 8);
        b0.q.lo = ldlds16(&tileB[buf][mrow][bkoff]);
        b0.q.hi = ldlds16(&tileB[buf][mrow][bkoff + 8]);
        b1.q.lo = ldlds16(&tileB[buf][16 + mrow][bkoff]);
        b1.q.hi = ldlds16(&tileB[buf][16 + mrow][bkoff + 8]);
        acc0 = wmma_bf16(a.v, b0.v, acc0);
        acc1 = wmma_bf16(a.v, b1.v, acc1);

        __syncthreads();
    }

    #pragma unroll
    for (int cf = 0; cf < 2; ++cf) {
        v8f acc = cf ? acc1 : acc0;
        const int n = nbase + cf * 16 + (lane & 15);
        const float bv = bias[n];
        #pragma unroll
        for (int r = 0; r < 8; ++r) {
            const int row = mbase + r + 8 * khalf;
            out[(size_t)row * ND + n] = acc[r] + bv;
        }
    }
}

// ---------------------------------------------------------------------------
extern "C" void kernel_launch(void* const* d_in, const int* in_sizes, int n_in,
                              void* d_out, int out_size, void* d_ws, size_t ws_size,
                              hipStream_t stream) {
    const float* x      = (const float*)d_in[0];
    const int*   tpos   = (const int*)  d_in[1];
    const float* Wqkv_w = (const float*)d_in[2];
    const float* Wqkv_b = (const float*)d_in[3];
    const float* Wo_w   = (const float*)d_in[4];
    const float* Wo_b   = (const float*)d_in[5];
    float* out = (float*)d_out;

    char* ws = (char*)d_ws;
    size_t off = 0;
    auto alloc = [&](size_t bytes) -> char* {
        char* p = ws + off;
        off += (bytes + 255) & ~(size_t)255;
        return p;
    };
    const size_t nX = (size_t)NB * NS * ND;
    bf16_t* xb    = (bf16_t*)alloc(nX * 2);
    bf16_t* wqkvb = (bf16_t*)alloc((size_t)N3D * ND * 2);
    bf16_t* wob   = (bf16_t*)alloc((size_t)ND * ND * 2);
    bf16_t* Qb    = (bf16_t*)alloc(nX * 2);
    bf16_t* Kb    = (bf16_t*)alloc(nX * 2);
    bf16_t* Vt    = (bf16_t*)alloc(nX * 2);
    bf16_t* AO    = (bf16_t*)alloc(nX * 2);

    cvt_f32_bf16<<<2048, 256, 0, stream>>>(x,      xb,    (int)nX);
    cvt_f32_bf16<<<2048, 256, 0, stream>>>(Wqkv_w, wqkvb, N3D * ND);
    cvt_f32_bf16<<<1024, 256, 0, stream>>>(Wo_w,   wob,   ND * ND);

    qkv_rope_kernel<<<dim3(N3D / 32, (NB * NS) / 64), 128, 0, stream>>>(
        xb, wqkvb, Wqkv_b, tpos, Qb, Kb, Vt);

    flash_attn_kernel<<<dim3(NS / 64, NB * NH), 128, 0, stream>>>(Qb, Kb, Vt, AO);

    oproj_kernel<<<dim3(ND / 32, (NB * NS) / 64), 128, 0, stream>>>(
        AO, wob, Wo_b, out);
}
